// LearnedImportanceTokenizer_5403068858480
// MI455X (gfx1250) — matrix operand order repaired and checked by reading, not compiled, not run on hardware
//
#include <hip/hip_runtime.h>
#include <hip/hip_bf16.h>

#define NPTS   131072
#define NBATCH 64
#define MAXPB  2560
#define MTOK   128
#define TOKD   768

typedef __bf16 bf16_t;
typedef bf16_t v16bf __attribute__((ext_vector_type(16)));
typedef bf16_t v8bf  __attribute__((ext_vector_type(8)));
typedef float  v8f   __attribute__((ext_vector_type(8)));

// ---------------------------------------------------------------------------
// Weight prep: W[K,Nw] f32 row-major  ->  WT[Nw,K] bf16 row-major
// ---------------------------------------------------------------------------
__global__ void wtrans_kernel(const float* __restrict__ W, bf16_t* __restrict__ WT,
                              int K, int Nw) {
    int idx = blockIdx.x * blockDim.x + threadIdx.x;
    if (idx >= K * Nw) return;
    int k = idx / Nw;
    int n = idx - k * Nw;
    WT[(size_t)n * K + k] = (bf16_t)W[idx];
}

// ---------------------------------------------------------------------------
// Per-point preprocessing: xyzt, LayerNorm(4), spatial MLP 4->64->64 (relu),
// emit h0 = [features(32) | spatial(64)] in bf16, and xyzt in f32.
// ---------------------------------------------------------------------------
__global__ __launch_bounds__(256)
void point_pre_kernel(const float* __restrict__ coords,
                      const float* __restrict__ features,
                      const float* __restrict__ times,
                      const float* __restrict__ ln_g, const float* __restrict__ ln_b,
                      const float* __restrict__ sw1,  const float* __restrict__ sb1,
                      const float* __restrict__ sw2,  const float* __restrict__ sb2,
                      float* __restrict__ xyzt_out,   bf16_t* __restrict__ h0) {
    __shared__ float s_sw1[4 * 64];
    __shared__ float s_sw2[64 * 64];
    __shared__ float s_sb1[64], s_sb2[64];
    __shared__ float s_g[4], s_b[4];
    for (int i = threadIdx.x; i < 4 * 64; i += blockDim.x)   s_sw1[i] = sw1[i];
    for (int i = threadIdx.x; i < 64 * 64; i += blockDim.x)  s_sw2[i] = sw2[i];
    if (threadIdx.x < 64) { s_sb1[threadIdx.x] = sb1[threadIdx.x];
                            s_sb2[threadIdx.x] = sb2[threadIdx.x]; }
    if (threadIdx.x < 4)  { s_g[threadIdx.x] = ln_g[threadIdx.x];
                            s_b[threadIdx.x] = ln_b[threadIdx.x]; }
    __syncthreads();

    int i = blockIdx.x * blockDim.x + threadIdx.x;
    if (i >= NPTS) return;

    float x[4] = { coords[3 * i], coords[3 * i + 1], coords[3 * i + 2], times[i] };
    float mu = 0.25f * (x[0] + x[1] + x[2] + x[3]);
    float var = 0.f;
    #pragma unroll
    for (int c = 0; c < 4; ++c) { float d = x[c] - mu; var += d * d; }
    var *= 0.25f;
    float inv = rsqrtf(var + 1e-5f);
    float xn[4];
    #pragma unroll
    for (int c = 0; c < 4; ++c) xn[c] = (x[c] - mu) * inv * s_g[c] + s_b[c];

    #pragma unroll
    for (int c = 0; c < 4; ++c) xyzt_out[4 * i + c] = x[c];

    float s1[64];
    #pragma unroll 8
    for (int j = 0; j < 64; ++j) {
        float a = s_sb1[j];
        #pragma unroll
        for (int c = 0; c < 4; ++c) a += xn[c] * s_sw1[c * 64 + j];
        s1[j] = fmaxf(a, 0.f);
    }

    bf16_t* out = h0 + (size_t)i * 96;
    const float* f = features + (size_t)i * 32;
    #pragma unroll 8
    for (int c = 0; c < 32; ++c) out[c] = (bf16_t)f[c];

    for (int j = 0; j < 64; ++j) {
        float a = s_sb2[j];
        #pragma unroll 8
        for (int c = 0; c < 64; ++c) a += s1[c] * s_sw2[c * 64 + j];
        out[32 + j] = (bf16_t)fmaxf(a, 0.f);
    }
}

// ---------------------------------------------------------------------------
// bf16 WMMA GEMM:  C[M,Nw] = act(A[M,K] @ W[K,Nw] + bias)
//   A  : bf16 row-major (K multiple of 32)
//   WT : bf16, W transposed -> [Nw, K] row-major (Nw multiple of 16)
//   Cb : bf16 out (optional), Cf : f32 out (optional)
// Block: 256 threads (8 waves). Block tile 256 rows x 64 cols.
// Each wave: 32-row strip (2 A frags), 8x (16x16) f32 accumulators.
// Software-pipelined: K-step n+1 fragments load while step n WMMAs issue.
// ---------------------------------------------------------------------------
struct FragSet { v16bf a0, a1, b0, b1, b2, b3; };

__device__ __forceinline__ v16bf pack16(v8bf lo, v8bf hi) {
    v16bf r;
    #pragma unroll
    for (int e = 0; e < 8; ++e) { r[e] = lo[e]; r[8 + e] = hi[e]; }
    return r;
}

__global__ __launch_bounds__(256)
void gemm_bf16_wmma(const bf16_t* __restrict__ A, const bf16_t* __restrict__ WT,
                    const float* __restrict__ bias,
                    bf16_t* __restrict__ Cb, float* __restrict__ Cf,
                    int K, int Nw, int do_relu) {
    const int lane = threadIdx.x & 31;
    const int wave = threadIdx.x >> 5;
    const int half = lane >> 4;     // 0: lanes 0-15, 1: lanes 16-31
    const int l16  = lane & 15;
    const int row_base = blockIdx.x * 256 + wave * 32;
    const int col_base = blockIdx.y * 64;

    // A fragment (16x32 bf16): lanes 0-15 -> K[0..7],K[16..23];
    // lanes 16-31 -> K[8..15],K[24..31]. Per lane: two contiguous 16B loads
    // at (k0 + half*8) and (k0 + half*8 + 16).
    const bf16_t* a0p = A + (size_t)(row_base + l16) * K + half * 8;
    const bf16_t* a1p = a0p + (size_t)16 * K;
    // B fragment (32x16 bf16): lane holds column, same K order as A:
    // two 16B loads at (k0 + half*8) and (k0 + half*8 + 16).
    const bf16_t* b0p = WT + (size_t)(col_base +  0 + l16) * K + half * 8;
    const bf16_t* b1p = WT + (size_t)(col_base + 16 + l16) * K + half * 8;
    const bf16_t* b2p = WT + (size_t)(col_base + 32 + l16) * K + half * 8;
    const bf16_t* b3p = WT + (size_t)(col_base + 48 + l16) * K + half * 8;

    v8f acc[8];
    #pragma unroll
    for (int j = 0; j < 8; ++j)
        #pragma unroll
        for (int e = 0; e < 8; ++e) acc[j][e] = 0.f;

    auto LOAD = [&](int k0, FragSet& f) {
        f.a0 = pack16(*(const v8bf*)(a0p + k0), *(const v8bf*)(a0p + k0 + 16));
        f.a1 = pack16(*(const v8bf*)(a1p + k0), *(const v8bf*)(a1p + k0 + 16));
        f.b0 = pack16(*(const v8bf*)(b0p + k0), *(const v8bf*)(b0p + k0 + 16));
        f.b1 = pack16(*(const v8bf*)(b1p + k0), *(const v8bf*)(b1p + k0 + 16));
        f.b2 = pack16(*(const v8bf*)(b2p + k0), *(const v8bf*)(b2p + k0 + 16));
        f.b3 = pack16(*(const v8bf*)(b3p + k0), *(const v8bf*)(b3p + k0 + 16));
    };
    auto MMA = [&](const FragSet& f) {
        acc[0] = __builtin_amdgcn_wmma_f32_16x16x32_bf16(false, f.a0, false, f.b0, (short)0, acc[0], false, false);
        acc[1] = __builtin_amdgcn_wmma_f32_16x16x32_bf16(false, f.a0, false, f.b1, (short)0, acc[1], false, false);
        acc[2] = __builtin_amdgcn_wmma_f32_16x16x32_bf16(false, f.a0, false, f.b2, (short)0, acc[2], false, false);
        acc[3] = __builtin_amdgcn_wmma_f32_16x16x32_bf16(false, f.a0, false, f.b3, (short)0, acc[3], false, false);
        acc[4] = __builtin_amdgcn_wmma_f32_16x16x32_bf16(false, f.a1, false, f.b0, (short)0, acc[4], false, false);
        acc[5] = __builtin_amdgcn_wmma_f32_16x16x32_bf16(false, f.a1, false, f.b1, (short)0, acc[5], false, false);
        acc[6] = __builtin_amdgcn_wmma_f32_16x16x32_bf16(false, f.a1, false, f.b2, (short)0, acc[6], false, false);
        acc[7] = __builtin_amdgcn_wmma_f32_16x16x32_bf16(false, f.a1, false, f.b3, (short)0, acc[7], false, false);
    };

    const int nsteps = K >> 5;          // K / 32, >= 2 for all layers here
    FragSet f0, f1;
    LOAD(0, f0);
    int step = 0;
    while (true) {
        if (step + 1 < nsteps) LOAD((step + 1) * 32, f1);
        MMA(f0);
        ++step;
        if (step >= nsteps) break;
        if (step + 1 < nsteps) LOAD((step + 1) * 32, f0);
        MMA(f1);
        ++step;
        if (step >= nsteps) break;
    }

    // Epilogue. C/D layout: VGPR v, lane -> (M = v + half*8, N = l16).
    #pragma unroll
    for (int s = 0; s < 2; ++s) {
        #pragma unroll
        for (int j = 0; j < 4; ++j) {
            const int col = col_base + j * 16 + l16;
            const float bv = bias ? bias[col] : 0.f;
            const v8f& av = acc[s * 4 + j];
            #pragma unroll
            for (int v = 0; v < 8; ++v) {
                const int row = row_base + s * 16 + half * 8 + v;
                float x = av[v] + bv;
                if (do_relu) x = fmaxf(x, 0.f);
                if (Cb) Cb[(size_t)row * Nw + col] = (bf16_t)x;
                if (Cf) Cf[(size_t)row * Nw + col] = x;
            }
        }
    }
}

// ---------------------------------------------------------------------------
// Final head: scores[i] = h5[i,:384] . hw2 + hb2
// ---------------------------------------------------------------------------
__global__ __launch_bounds__(256)
void head2_kernel(const bf16_t* __restrict__ h5, const float* __restrict__ hw2,
                  const float* __restrict__ hb2, float* __restrict__ scores) {
    __shared__ float w[384];
    for (int i = threadIdx.x; i < 384; i += blockDim.x) w[i] = hw2[i];
    __syncthreads();
    int i = blockIdx.x * blockDim.x + threadIdx.x;
    if (i >= NPTS) return;
    const bf16_t* h = h5 + (size_t)i * 384;
    float s = hb2[0];
    #pragma unroll 8
    for (int j = 0; j < 384; ++j) s += (float)h[j] * w[j];
    scores[i] = s;
}

// ---------------------------------------------------------------------------
// Per-batch top-128 (jax.lax.top_k semantics: descending, ties -> lower index)
// + hard gather of tokens / centroids / mask. One block per batch.
// batch_ids is sorted -> each batch is a contiguous [off, off+cnt) range.
// ---------------------------------------------------------------------------
__global__ __launch_bounds__(256)
void topk_gather_kernel(const int* __restrict__ batch_ids,
                        const float* __restrict__ scores,
                        const float* __restrict__ pf,     // point_feats f32 [N,768]
                        const float* __restrict__ xyzt,   // [N,4]
                        float* __restrict__ out_tokens,   // [B,128,768]
                        float* __restrict__ out_cent,     // [B,128,4]
                        float* __restrict__ out_mask) {   // [B,128]
    __shared__ float S[MAXPB];
    __shared__ float rv[256];
    __shared__ int   ri[256];
    __shared__ int   s_top[MTOK];
    __shared__ int   s_off, s_cnt;

    const int b = blockIdx.x;
    if (threadIdx.x == 0) {
        int lo = 0, hi = NPTS;
        while (lo < hi) { int m = (lo + hi) >> 1; if (batch_ids[m] < b) lo = m + 1; else hi = m; }
        int off = lo;
        lo = off; hi = NPTS;
        while (lo < hi) { int m = (lo + hi) >> 1; if (batch_ids[m] < b + 1) lo = m + 1; else hi = m; }
        s_off = off;
        s_cnt = lo - off;
    }
    __syncthreads();
    const int off = s_off;
    int cnt = s_cnt;
    if (cnt > MAXPB) cnt = MAXPB;   // reference scatter drops pos >= MAX_PER_B

    const float NEG_MAX = -__FLT_MAX__;        // reference NEG_INF = finfo.min
    for (int p = threadIdx.x; p < MAXPB; p += 256)
        S[p] = (p < cnt) ? scores[off + p] : NEG_MAX;
    __syncthreads();

    for (int k = 0; k < MTOK; ++k) {
        float bv = -__builtin_inff();          // below NEG_MAX -> pads outrank consumed
        int   bi = 1 << 30;
        for (int p = threadIdx.x; p < MAXPB; p += 256) {
            float v = S[p];
            if (v > bv) { bv = v; bi = p; }    // strict > keeps smallest index
        }
        rv[threadIdx.x] = bv; ri[threadIdx.x] = bi;
        __syncthreads();
        for (int sd = 128; sd > 0; sd >>= 1) {
            if (threadIdx.x < sd) {
                float v2 = rv[threadIdx.x + sd];
                int   i2 = ri[threadIdx.x + sd];
                if (v2 > rv[threadIdx.x] ||
                    (v2 == rv[threadIdx.x] && i2 < ri[threadIdx.x])) {
                    rv[threadIdx.x] = v2; ri[threadIdx.x] = i2;
                }
            }
            __syncthreads();
        }
        if (threadIdx.x == 0) { s_top[k] = ri[0]; S[ri[0]] = -__builtin_inff(); }
        __syncthreads();
    }

    // Gather tokens (zero for invalid, matching point_feats_pad[N] == 0)
    for (int idx = threadIdx.x; idx < MTOK * TOKD; idx += 256) {
        int t = idx / TOKD, c = idx - t * TOKD;
        int pos = s_top[t];
        bool valid = pos < cnt;
        out_tokens[(size_t)b * MTOK * TOKD + idx] =
            valid ? pf[(size_t)(off + pos) * TOKD + c] : 0.f;
    }
    for (int idx = threadIdx.x; idx < MTOK * 4; idx += 256) {
        int t = idx >> 2, c = idx & 3;
        int pos = s_top[t];
        bool valid = pos < cnt;
        out_cent[(size_t)b * MTOK * 4 + idx] =
            valid ? xyzt[(size_t)(off + pos) * 4 + c] : 0.f;
    }
    if (threadIdx.x < MTOK)
        out_mask[b * MTOK + threadIdx.x] = (s_top[threadIdx.x] < cnt) ? 1.f : 0.f;
}

// ---------------------------------------------------------------------------
// Launch
// ---------------------------------------------------------------------------
extern "C" void kernel_launch(void* const* d_in, const int* in_sizes, int n_in,
                              void* d_out, int out_size, void* d_ws, size_t ws_size,
                              hipStream_t stream) {
    const float* coords   = (const float*)d_in[0];
    const float* features = (const float*)d_in[1];
    const float* times    = (const float*)d_in[2];
    const int*   bids     = (const int*)  d_in[3];
    const float* ln_g = (const float*)d_in[4];
    const float* ln_b = (const float*)d_in[5];
    const float* sw1  = (const float*)d_in[6];
    const float* sb1  = (const float*)d_in[7];
    const float* sw2  = (const float*)d_in[8];
    const float* sb2  = (const float*)d_in[9];
    const float* mw1  = (const float*)d_in[10];
    const float* mb1  = (const float*)d_in[11];
    const float* mw2  = (const float*)d_in[12];
    const float* mb2  = (const float*)d_in[13];
    const float* mw3  = (const float*)d_in[14];
    const float* mb3  = (const float*)d_in[15];
    const float* mw4  = (const float*)d_in[16];
    const float* mb4  = (const float*)d_in[17];
    const float* hw1  = (const float*)d_in[18];
    const float* hb1  = (const float*)d_in[19];
    const float* hw2  = (const float*)d_in[20];
    const float* hb2  = (const float*)d_in[21];

    // Workspace layout
    char* ws = (char*)d_ws;
    size_t o = 0;
    float*  pf    = (float*)(ws + o);  o += (size_t)NPTS * TOKD * 4;   // 402 MB
    float*  xyzt  = (float*)(ws + o);  o += (size_t)NPTS * 4 * 4;
    float*  scr   = (float*)(ws + o);  o += (size_t)NPTS * 4;
    bf16_t* bufA  = (bf16_t*)(ws + o); o += (size_t)NPTS * TOKD * 2;   // 201 MB
    bf16_t* bufB  = (bf16_t*)(ws + o); o += (size_t)NPTS * TOKD * 2;   // 201 MB
    bf16_t* WT1   = (bf16_t*)(ws + o); o += (size_t)96  * 256 * 2;
    bf16_t* WT2   = (bf16_t*)(ws + o); o += (size_t)256 * 512 * 2;
    bf16_t* WT3   = (bf16_t*)(ws + o); o += (size_t)512 * 768 * 2;
    bf16_t* WT4   = (bf16_t*)(ws + o); o += (size_t)768 * 768 * 2;
    bf16_t* WTh   = (bf16_t*)(ws + o); o += (size_t)768 * 384 * 2;
    if (o > ws_size) return;   // workspace too small: bail deterministically

    float* out_tokens = (float*)d_out;
    float* out_cent   = out_tokens + (size_t)NBATCH * MTOK * TOKD;
    float* out_mask   = out_cent   + (size_t)NBATCH * MTOK * 4;

    // 1) weight transpose + bf16 convert
    wtrans_kernel<<<( 96*256 + 255)/256, 256, 0, stream>>>(mw1, WT1,  96, 256);
    wtrans_kernel<<<(256*512 + 255)/256, 256, 0, stream>>>(mw2, WT2, 256, 512);
    wtrans_kernel<<<(512*768 + 255)/256, 256, 0, stream>>>(mw3, WT3, 512, 768);
    wtrans_kernel<<<(768*768 + 255)/256, 256, 0, stream>>>(mw4, WT4, 768, 768);
    wtrans_kernel<<<(768*384 + 255)/256, 256, 0, stream>>>(hw1, WTh, 768, 384);

    // 2) preprocessing + spatial MLP -> h0 (bf16, N x 96) in bufA
    point_pre_kernel<<<NPTS / 256, 256, 0, stream>>>(
        coords, features, times, ln_g, ln_b, sw1, sb1, sw2, sb2, xyzt, bufA);

    // 3) GEMM chain (WMMA bf16), block tile 256 rows x 64 cols
    const int GX = NPTS / 256;
    gemm_bf16_wmma<<<dim3(GX, 256 / 64), 256, 0, stream>>>(bufA, WT1, mb1, bufB, nullptr,  96, 256, 1);
    gemm_bf16_wmma<<<dim3(GX, 512 / 64), 256, 0, stream>>>(bufB, WT2, mb2, bufA, nullptr, 256, 512, 1);
    gemm_bf16_wmma<<<dim3(GX, 768 / 64), 256, 0, stream>>>(bufA, WT3, mb3, bufB, nullptr, 512, 768, 1);
    gemm_bf16_wmma<<<dim3(GX, 768 / 64), 256, 0, stream>>>(bufB, WT4, mb4, bufA, pf,      768, 768, 0);
    gemm_bf16_wmma<<<dim3(GX, 384 / 64), 256, 0, stream>>>(bufA, WTh, hb1, bufB, nullptr, 768, 384, 1);

    // 4) scores
    head2_kernel<<<NPTS / 256, 256, 0, stream>>>(bufB, hw2, hb2, scr);

    // 5) per-batch top-128 + gather
    topk_gather_kernel<<<NBATCH, 256, 0, stream>>>(
        bids, scr, pf, xyzt, out_tokens, out_cent, out_mask);
}
